// DKVMN_53961969107221
// MI455X (gfx1250) — matrix operand order repaired and checked by reading, not compiled, not run on hardware
//
#include <hip/hip_runtime.h>

// ---------------- problem constants ----------------
#define NUM_C    100
#define DIM_S    128
#define SIZE_M   64
#define EMB      128
#define PLEN     128                 // MAX_CODE_LEN
#define TBL_ROWS 10002               // NODE_COUNT+2 == PATH_COUNT+2
#define VEMB_ROWS 202                // NUM_C*2 + 2
#define KEMB_ROWS 101                // NUM_C + 1
#define BSZ      16
#define LSEQ     128
#define CODE_W   (NUM_C * 2 + PLEN * 3)   // 584
#define K3E      (3 * EMB)                // 384
#define ROWS_BIG (BSZ * LSEQ * PLEN)      // 262144
#define ROWS_SEQ (BSZ * LSEQ)             // 2048

// ---------------- WMMA types ----------------
typedef __attribute__((ext_vector_type(16))) __bf16 v16bf;
typedef __attribute__((ext_vector_type(8)))  float  v8f;

union BF16x16 { v16bf v; uint4 q[2]; };

__device__ __forceinline__ unsigned short f2bf(float f) {
    unsigned u = __float_as_uint(f);
    u += 0x7FFFu + ((u >> 16) & 1u);          // round-to-nearest-even
    return (unsigned short)(u >> 16);
}

// stable fast tanh: 1 - 2/(e^{2x}+1)  (inf-safe, 2 transcendentals)
__device__ __forceinline__ float fast_tanh(float x) {
    return 1.f - 2.f / (__expf(2.f * x) + 1.f);
}

__device__ __forceinline__ v8f wmma_bf16(const BF16x16& a, const BF16x16& b, v8f c) {
    return __builtin_amdgcn_wmma_f32_16x16x32_bf16(
        false, a.v, false, b.v, (short)0, c, false, false);
}

// ==================================================================
// Prep: f32 -> bf16 conversions (tables + transposed weights)
// ==================================================================
__global__ void cvt_bf16_kernel(const float* __restrict__ src,
                                unsigned short* __restrict__ dst, int n) {
    int i = blockIdx.x * blockDim.x + threadIdx.x;
    if (i < n) dst[i] = f2bf(src[i]);
}

// WT[n*K + k] = W[k*N + n]  (bf16)
__global__ void transpose_cvt_kernel(const float* __restrict__ W,
                                     unsigned short* __restrict__ WT, int K, int N) {
    int i = blockIdx.x * blockDim.x + threadIdx.x;
    if (i >= K * N) return;
    int n = i / K, k = i - n * K;
    WT[n * K + k] = f2bf(W[k * N + n]);
}

// WeaT[n*512 + k] = (n<128 ? W_e[k*128+n] : W_a[k*128+(n-128)])
__global__ void build_wea_kernel(const float* __restrict__ We,
                                 const float* __restrict__ Wa,
                                 unsigned short* __restrict__ WT) {
    int i = blockIdx.x * blockDim.x + threadIdx.x;
    if (i >= 256 * 512) return;
    int n = i >> 9, k = i & 511;
    float f = (n < DIM_S) ? We[k * DIM_S + n] : Wa[k * DIM_S + (n - DIM_S)];
    WT[n * 512 + k] = f2bf(f);
}

// ==================================================================
// K1: cw[row] = sum_n tanh((full@Wpt)[row,n] + b_pt[n]) * W_att[n] + b_att
//     64 rows / WG (4 M-blocks), 8 waves x 3 N-tiles, B reused 4x per load
// ==================================================================
#define K1_STRIDE 392   // 384 + 8 ushorts: conflict-free ds_load_b128, 16B aligned

__global__ __launch_bounds__(256) void big_gemm_cw_kernel(
    const int* __restrict__ code,
    const unsigned short* __restrict__ nodes_bf,
    const unsigned short* __restrict__ paths_bf,
    const unsigned short* __restrict__ WptT,     // [384][384] bf16 (N-major)
    const float* __restrict__ b_pt,
    const float* __restrict__ W_att,
    const float* __restrict__ b_att,
    float* __restrict__ cw)
{
    __shared__ __align__(16) unsigned short Abuf[64 * K1_STRIDE];
    __shared__ float cw_s[64];

    const int tid  = threadIdx.x;
    const int row0 = blockIdx.x * 64;

    // vectorized gather: 64 rows x 48 uint4-chunks (3 segments of 256B per row)
    for (int idx = tid; idx < 64 * 48; idx += 256) {
        int rl = idx / 48, cc = idx - rl * 48;
        int row = row0 + rl;
        int p   = row & (PLEN - 1);
        int bl  = row >> 7;                       // b*L + l
        const int* cp = code + bl * CODE_W + 2 * NUM_C + 3 * p;
        const uint4* src;
        if (cc < 16)       src = (const uint4*)(nodes_bf + cp[0] * EMB) + cc;        // start
        else if (cc < 32)  src = (const uint4*)(nodes_bf + cp[2] * EMB) + (cc - 16); // end
        else               src = (const uint4*)(paths_bf + cp[1] * EMB) + (cc - 32); // path
        *(uint4*)(Abuf + rl * K1_STRIDE + cc * 8) = *src;
    }
    if (tid < 64) cw_s[tid] = 0.f;
    __syncthreads();

    const int wave = tid >> 5;
    const int lane = tid & 31;
    const int m16  = lane & 15;
    const int hi   = lane >> 4;

    const int nb  = wave * 48 + m16;
    const int bo0 = (nb +  0) * K3E + hi * 16;
    const int bo1 = (nb + 16) * K3E + hi * 16;
    const int bo2 = (nb + 32) * K3E + hi * 16;

    const unsigned short* Ap0 = Abuf + ( 0 + m16) * K1_STRIDE + hi * 16;
    const unsigned short* Ap1 = Abuf + (16 + m16) * K1_STRIDE + hi * 16;
    const unsigned short* Ap2 = Abuf + (32 + m16) * K1_STRIDE + hi * 16;
    const unsigned short* Ap3 = Abuf + (48 + m16) * K1_STRIDE + hi * 16;

    v8f acc[4][3];
#pragma unroll
    for (int mb = 0; mb < 4; mb++)
#pragma unroll
        for (int i = 0; i < 3; i++) acc[mb][i] = (v8f){};

    for (int k0 = 0; k0 < K3E; k0 += 32) {
        BF16x16 b0, b1, b2;
        b0.q[0] = *(const uint4*)(WptT + bo0 + k0);  b0.q[1] = *(const uint4*)(WptT + bo0 + k0 + 8);
        b1.q[0] = *(const uint4*)(WptT + bo1 + k0);  b1.q[1] = *(const uint4*)(WptT + bo1 + k0 + 8);
        b2.q[0] = *(const uint4*)(WptT + bo2 + k0);  b2.q[1] = *(const uint4*)(WptT + bo2 + k0 + 8);

        BF16x16 a0, a1, a2, a3;
        a0.q[0] = *(const uint4*)(Ap0 + k0);  a0.q[1] = *(const uint4*)(Ap0 + k0 + 8);
        a1.q[0] = *(const uint4*)(Ap1 + k0);  a1.q[1] = *(const uint4*)(Ap1 + k0 + 8);
        a2.q[0] = *(const uint4*)(Ap2 + k0);  a2.q[1] = *(const uint4*)(Ap2 + k0 + 8);
        a3.q[0] = *(const uint4*)(Ap3 + k0);  a3.q[1] = *(const uint4*)(Ap3 + k0 + 8);

        acc[0][0] = wmma_bf16(a0, b0, acc[0][0]);
        acc[0][1] = wmma_bf16(a0, b1, acc[0][1]);
        acc[0][2] = wmma_bf16(a0, b2, acc[0][2]);
        acc[1][0] = wmma_bf16(a1, b0, acc[1][0]);
        acc[1][1] = wmma_bf16(a1, b1, acc[1][1]);
        acc[1][2] = wmma_bf16(a1, b2, acc[1][2]);
        acc[2][0] = wmma_bf16(a2, b0, acc[2][0]);
        acc[2][1] = wmma_bf16(a2, b1, acc[2][1]);
        acc[2][2] = wmma_bf16(a2, b2, acc[2][2]);
        acc[3][0] = wmma_bf16(a3, b0, acc[3][0]);
        acc[3][1] = wmma_bf16(a3, b1, acc[3][1]);
        acc[3][2] = wmma_bf16(a3, b2, acc[3][2]);
    }

    // epilogue: tanh(+bias)*W_att, butterfly-reduce over the 16 lanes per row
    const float bp0 = b_pt[nb], bp1 = b_pt[nb + 16], bp2 = b_pt[nb + 32];
    const float wa0 = W_att[nb], wa1 = W_att[nb + 16], wa2 = W_att[nb + 32];
#pragma unroll
    for (int mb = 0; mb < 4; mb++) {
#pragma unroll
        for (int v = 0; v < 8; v++) {
            float s = fast_tanh(acc[mb][0][v] + bp0) * wa0
                    + fast_tanh(acc[mb][1][v] + bp1) * wa1
                    + fast_tanh(acc[mb][2][v] + bp2) * wa2;
            s += __shfl_xor(s, 1, 32);
            s += __shfl_xor(s, 2, 32);
            s += __shfl_xor(s, 4, 32);
            s += __shfl_xor(s, 8, 32);
            if (m16 == 0) atomicAdd(&cw_s[mb * 16 + v + hi * 8], s);
        }
    }
    __syncthreads();
    if (tid < 64) cw[row0 + tid] = cw_s[tid] + b_att[0];
}

// ==================================================================
// K2: softmax over L (axis=1) for each (b, p)
// ==================================================================
__global__ __launch_bounds__(128) void softmax_L_kernel(float* __restrict__ cw) {
    __shared__ float red[128];
    const int bp = blockIdx.x;          // b*PLEN + p
    const int b  = bp >> 7;
    const int p  = bp & 127;
    const int l  = threadIdx.x;
    const int idx = (b * LSEQ + l) * PLEN + p;

    float x = cw[idx];
    red[l] = x; __syncthreads();
    for (int s = 64; s > 0; s >>= 1) { if (l < s) red[l] = fmaxf(red[l], red[l + s]); __syncthreads(); }
    float mx = red[0]; __syncthreads();
    float ex = __expf(x - mx);
    red[l] = ex; __syncthreads();
    for (int s = 64; s > 0; s >>= 1) { if (l < s) red[l] += red[l + s]; __syncthreads(); }
    cw[idx] = ex / red[0];
}

// ==================================================================
// K3: code_vec[bl, e] = sum_p attn[bl,p] * full[bl,p,e]  -> bf16 output
// ==================================================================
__global__ __launch_bounds__(384) void codevec_kernel(
    const int* __restrict__ code, const float* __restrict__ attn,
    const float* __restrict__ nodes, const float* __restrict__ paths,
    unsigned short* __restrict__ cvec_bf)
{
    __shared__ float att_s[PLEN];
    __shared__ int   idx_s[PLEN * 3];
    const int bl = blockIdx.x;
    const int e  = threadIdx.x;

    for (int i = e; i < PLEN; i += 384)      att_s[i] = attn[bl * PLEN + i];
    for (int i = e; i < PLEN * 3; i += 384)  idx_s[i] = code[bl * CODE_W + 2 * NUM_C + i];
    __syncthreads();

    float acc = 0.f;
    if (e < EMB) {
        for (int p = 0; p < PLEN; p++) acc += att_s[p] * nodes[idx_s[3 * p] * EMB + e];
    } else if (e < 2 * EMB) {
        const int e2 = e - EMB;
        for (int p = 0; p < PLEN; p++) acc += att_s[p] * nodes[idx_s[3 * p + 2] * EMB + e2];
    } else {
        const int e2 = e - 2 * EMB;
        for (int p = 0; p < PLEN; p++) acc += att_s[p] * paths[idx_s[3 * p + 1] * EMB + e2];
    }
    cvec_bf[bl * K3E + e] = f2bf(acc);
}

// ==================================================================
// K4a: w = softmax(k @ Mk^T) over M=64 via WMMA; 32 rows / WG
//      Note B[k][n] = Mk[n][k] -> row-major Mk is already the B layout.
// ==================================================================
#define KW_STRIDE 136   // 128 + 8

__global__ __launch_bounds__(256) void wsoft_gemm_kernel(
    const int* __restrict__ q, const unsigned short* __restrict__ kemb_bf,
    const unsigned short* __restrict__ Mk_bf, float* __restrict__ w)
{
    __shared__ __align__(16) unsigned short Abuf[32 * KW_STRIDE];
    __shared__ float dot_s[32 * 65];
    const int tid  = threadIdx.x;
    const int row0 = blockIdx.x * 32;

    for (int idx = tid; idx < 32 * 16; idx += 256) {
        int rl = idx >> 4, cc = idx & 15;
        *(uint4*)(Abuf + rl * KW_STRIDE + cc * 8) =
            *((const uint4*)(kemb_bf + q[row0 + rl] * DIM_S) + cc);
    }
    __syncthreads();

    const int wave = tid >> 5;
    const int lane = tid & 31;
    const int m16  = lane & 15;
    const int hi   = lane >> 4;
    const int mb   = wave >> 2;            // 0..1 (M-block)
    const int nt   = wave & 3;             // 0..3 (N-tile)
    const int nIdx = nt * 16 + m16;
    const int boB  = nIdx * DIM_S + hi * 16;
    const unsigned short* Ap = Abuf + (mb * 16 + m16) * KW_STRIDE + hi * 16;

    v8f acc = {};
#pragma unroll
    for (int k0 = 0; k0 < DIM_S; k0 += 32) {
        BF16x16 a, b;
        a.q[0] = *(const uint4*)(Ap + k0);        a.q[1] = *(const uint4*)(Ap + k0 + 8);
        b.q[0] = *(const uint4*)(Mk_bf + boB + k0); b.q[1] = *(const uint4*)(Mk_bf + boB + k0 + 8);
        acc = wmma_bf16(a, b, acc);
    }
#pragma unroll
    for (int v = 0; v < 8; v++)
        dot_s[(mb * 16 + v + hi * 8) * 65 + nIdx] = acc[v];
    __syncthreads();

    // softmax over 64 per row: 8 lanes per row, shfl butterfly
    const int row = tid >> 3;
    const int sub = tid & 7;
    float v8[8], mx = -3.4e38f;
#pragma unroll
    for (int i = 0; i < 8; i++) { v8[i] = dot_s[row * 65 + sub * 8 + i]; mx = fmaxf(mx, v8[i]); }
    mx = fmaxf(mx, __shfl_xor(mx, 1, 32));
    mx = fmaxf(mx, __shfl_xor(mx, 2, 32));
    mx = fmaxf(mx, __shfl_xor(mx, 4, 32));
    float sum = 0.f;
#pragma unroll
    for (int i = 0; i < 8; i++) { v8[i] = __expf(v8[i] - mx); sum += v8[i]; }
    sum += __shfl_xor(sum, 1, 32);
    sum += __shfl_xor(sum, 2, 32);
    sum += __shfl_xor(sum, 4, 32);
    const float inv = 1.f / sum;
#pragma unroll
    for (int i = 0; i < 8; i++)
        w[(row0 + row) * SIZE_M + sub * 8 + i] = v8[i] * inv;
}

// ==================================================================
// K4: e = sigmoid(v@W_e+b_e), a = tanh(v@W_a+b_a); v = [vr | code_vec] (K=512)
// ==================================================================
#define K4_STRIDE 520   // 512 + 8 ushorts

__global__ __launch_bounds__(256) void ea_gemm_kernel(
    const int* __restrict__ r, const unsigned short* __restrict__ vemb_bf,
    const unsigned short* __restrict__ cvec_bf, const unsigned short* __restrict__ WeaT,
    const float* __restrict__ b_e, const float* __restrict__ b_a,
    float* __restrict__ e_out, float* __restrict__ a_out)
{
    __shared__ __align__(16) unsigned short Abuf[16 * K4_STRIDE];
    const int tid  = threadIdx.x;
    const int row0 = blockIdx.x * 16;

    for (int idx = tid; idx < 16 * 64; idx += 256) {
        int rl = idx >> 6, cc = idx & 63;
        int row = row0 + rl;
        const uint4* src = (cc < 16)
            ? (const uint4*)(vemb_bf + r[row] * DIM_S) + cc
            : (const uint4*)(cvec_bf + row * K3E) + (cc - 16);
        *(uint4*)(Abuf + rl * K4_STRIDE + cc * 8) = *src;
    }
    __syncthreads();

    const int wave = tid >> 5;
    const int lane = tid & 31;
    const int m16  = lane & 15;
    const int hi   = lane >> 4;

    const int nA = wave * 32 + m16;
    const int nB = wave * 32 + 16 + m16;
    const int boA = nA * 512 + hi * 16;
    const int boB = nB * 512 + hi * 16;
    const unsigned short* Ap = Abuf + m16 * K4_STRIDE + hi * 16;

    v8f acc0 = {}, acc1 = {};
    for (int k0 = 0; k0 < 512; k0 += 32) {
        BF16x16 a;
        a.q[0] = *(const uint4*)(Ap + k0);
        a.q[1] = *(const uint4*)(Ap + k0 + 8);
        BF16x16 b0, b1;
        b0.q[0] = *(const uint4*)(WeaT + boA + k0);  b0.q[1] = *(const uint4*)(WeaT + boA + k0 + 8);
        b1.q[0] = *(const uint4*)(WeaT + boB + k0);  b1.q[1] = *(const uint4*)(WeaT + boB + k0 + 8);
        acc0 = wmma_bf16(a, b0, acc0);
        acc1 = wmma_bf16(a, b1, acc1);
    }

    const float bias0 = (nA < DIM_S) ? b_e[nA] : b_a[nA - DIM_S];
    const float bias1 = (nB < DIM_S) ? b_e[nB] : b_a[nB - DIM_S];
    const int moff = hi * 8;
#pragma unroll
    for (int v = 0; v < 8; v++) {
        int row = row0 + v + moff;
        float x0 = acc0[v] + bias0;
        float x1 = acc1[v] + bias1;
        if (nA < DIM_S) e_out[row * DIM_S + nA] = 1.f / (1.f + __expf(-x0));
        else            a_out[row * DIM_S + (nA - DIM_S)] = fast_tanh(x0);
        if (nB < DIM_S) e_out[row * DIM_S + nB] = 1.f / (1.f + __expf(-x1));
        else            a_out[row * DIM_S + (nB - DIM_S)] = fast_tanh(x1);
    }
}

// ==================================================================
// K5: sequential DKVMN scan over L; thread d owns register column Mv[:,d]
// ==================================================================
__global__ __launch_bounds__(128) void scan_kernel(
    const float* __restrict__ Mv0, const float* __restrict__ w,
    const float* __restrict__ e, const float* __restrict__ a,
    float* __restrict__ readout)
{
    __shared__ float w_s[SIZE_M];
    const int b = blockIdx.x;
    const int d = threadIdx.x;

    float mv[SIZE_M];
#pragma unroll
    for (int m = 0; m < SIZE_M; m++) mv[m] = Mv0[m * DIM_S + d];

    for (int l = 0; l < LSEQ; l++) {
        const int row = b * LSEQ + l;
        if (d < SIZE_M) w_s[d] = w[row * SIZE_M + d];
        __syncthreads();
        const float ed = e[row * DIM_S + d];
        const float ad = a[row * DIM_S + d];
        float r0 = 0.f, r1 = 0.f, r2 = 0.f, r3 = 0.f;
#pragma unroll
        for (int m = 0; m < SIZE_M; m += 4) {
            r0 += w_s[m + 0] * mv[m + 0];
            r1 += w_s[m + 1] * mv[m + 1];
            r2 += w_s[m + 2] * mv[m + 2];
            r3 += w_s[m + 3] * mv[m + 3];
        }
        readout[row * DIM_S + d] = (r0 + r1) + (r2 + r3);
#pragma unroll
        for (int m = 0; m < SIZE_M; m++) {
            const float wm = w_s[m];
            mv[m] = mv[m] * (1.f - wm * ed) + wm * ad;
        }
        __syncthreads();
    }
}

// ==================================================================
// K6: f = tanh([read|k]@W_f+b_f) via WMMA (16 rows/WG), then out = sigmoid(f@W_p+b_p)
// ==================================================================
#define K6_STRIDE 264   // 256 + 8

__global__ __launch_bounds__(256) void out_gemm_kernel(
    const float* __restrict__ readout, const int* __restrict__ q,
    const unsigned short* __restrict__ kemb_bf,
    const unsigned short* __restrict__ WfT,       // [128][256] bf16 (N-major)
    const float* __restrict__ b_f, const float* __restrict__ W_p,
    const float* __restrict__ b_p, float* __restrict__ out)
{
    __shared__ __align__(16) unsigned short Abuf[16 * K6_STRIDE];
    __shared__ float f_s[16][132];
    const int tid  = threadIdx.x;
    const int row0 = blockIdx.x * 16;

    for (int idx = tid; idx < 16 * 256; idx += 256) {
        int rl = idx >> 8, j = idx & 255;
        int row = row0 + rl;
        unsigned short val = (j < DIM_S) ? f2bf(readout[row * DIM_S + j])
                                         : kemb_bf[q[row] * DIM_S + (j - DIM_S)];
        Abuf[rl * K6_STRIDE + j] = val;
    }
    __syncthreads();

    const int wave = tid >> 5;
    const int lane = tid & 31;
    const int m16  = lane & 15;
    const int hi   = lane >> 4;
    const int nIdx = wave * 16 + m16;
    const int boB  = nIdx * 256 + hi * 16;
    const unsigned short* Ap = Abuf + m16 * K6_STRIDE + hi * 16;

    v8f acc = {};
#pragma unroll
    for (int k0 = 0; k0 < 256; k0 += 32) {
        BF16x16 a, b;
        a.q[0] = *(const uint4*)(Ap + k0);         a.q[1] = *(const uint4*)(Ap + k0 + 8);
        b.q[0] = *(const uint4*)(WfT + boB + k0);  b.q[1] = *(const uint4*)(WfT + boB + k0 + 8);
        acc = wmma_bf16(a, b, acc);
    }
    const float bf = b_f[nIdx];
#pragma unroll
    for (int v = 0; v < 8; v++)
        f_s[v + hi * 8][nIdx] = fast_tanh(acc[v] + bf);
    __syncthreads();

    if (tid < 160) {
        const int rl = tid / 10;
        const int o  = tid - rl * 10;
        float oacc = b_p[o];
        for (int dd = 0; dd < DIM_S; dd++) oacc += f_s[rl][dd] * W_p[dd * 10 + o];
        out[(row0 + rl) * 10 + o] = 1.f / (1.f + __expf(-oacc));
    }
}

// ==================================================================
// Host launcher
// ==================================================================
extern "C" void kernel_launch(void* const* d_in, const int* in_sizes, int n_in,
                              void* d_out, int out_size, void* d_ws, size_t ws_size,
                              hipStream_t stream) {
    const int*   code   = (const int*)  d_in[0];
    const int*   q      = (const int*)  d_in[1];
    const int*   r      = (const int*)  d_in[2];
    const float* nodes  = (const float*)d_in[3];
    const float* paths  = (const float*)d_in[4];
    const float* W_pt   = (const float*)d_in[5];
    const float* b_pt   = (const float*)d_in[6];
    const float* W_att  = (const float*)d_in[7];
    const float* b_att  = (const float*)d_in[8];
    const float* k_emb  = (const float*)d_in[9];
    const float* Mk     = (const float*)d_in[10];
    const float* Mv0    = (const float*)d_in[11];
    const float* v_emb  = (const float*)d_in[12];
    const float* W_e    = (const float*)d_in[13];
    const float* b_e    = (const float*)d_in[14];
    const float* W_a    = (const float*)d_in[15];
    const float* b_a    = (const float*)d_in[16];
    const float* W_f    = (const float*)d_in[17];
    const float* b_f    = (const float*)d_in[18];
    const float* W_p    = (const float*)d_in[19];
    const float* b_p    = (const float*)d_in[20];
    float* out = (float*)d_out;

    // workspace carve-up (256B aligned)
    size_t off = 0;
    auto carve = [&](size_t bytes) -> char* {
        char* p = (char*)d_ws + off;
        off += (bytes + 255) & ~(size_t)255;
        return p;
    };
    unsigned short* nodes_bf = (unsigned short*)carve((size_t)TBL_ROWS * EMB * 2);
    unsigned short* paths_bf = (unsigned short*)carve((size_t)TBL_ROWS * EMB * 2);
    unsigned short* WptT     = (unsigned short*)carve((size_t)K3E * K3E * 2);
    unsigned short* WeaT     = (unsigned short*)carve((size_t)256 * 512 * 2);
    unsigned short* WfT      = (unsigned short*)carve((size_t)128 * 256 * 2);
    unsigned short* vemb_bf  = (unsigned short*)carve((size_t)VEMB_ROWS * DIM_S * 2);
    unsigned short* kemb_bf  = (unsigned short*)carve((size_t)KEMB_ROWS * DIM_S * 2);
    unsigned short* Mk_bf    = (unsigned short*)carve((size_t)SIZE_M * DIM_S * 2);
    float* cw       = (float*)carve((size_t)ROWS_BIG * 4);
    unsigned short* cvec_bf = (unsigned short*)carve((size_t)ROWS_SEQ * K3E * 2);
    float* w_ws     = (float*)carve((size_t)ROWS_SEQ * SIZE_M * 4);
    float* e_ws     = (float*)carve((size_t)ROWS_SEQ * DIM_S * 4);
    float* a_ws     = (float*)carve((size_t)ROWS_SEQ * DIM_S * 4);
    float* read_ws  = (float*)carve((size_t)ROWS_SEQ * DIM_S * 4);
    (void)ws_size; (void)n_in; (void)in_sizes; (void)out_size;

    const int ntbl = TBL_ROWS * EMB;
    cvt_bf16_kernel<<<(ntbl + 255) / 256, 256, 0, stream>>>(nodes, nodes_bf, ntbl);
    cvt_bf16_kernel<<<(ntbl + 255) / 256, 256, 0, stream>>>(paths, paths_bf, ntbl);
    cvt_bf16_kernel<<<(VEMB_ROWS * DIM_S + 255) / 256, 256, 0, stream>>>(v_emb, vemb_bf, VEMB_ROWS * DIM_S);
    cvt_bf16_kernel<<<(KEMB_ROWS * DIM_S + 255) / 256, 256, 0, stream>>>(k_emb, kemb_bf, KEMB_ROWS * DIM_S);
    cvt_bf16_kernel<<<(SIZE_M * DIM_S + 255) / 256, 256, 0, stream>>>(Mk, Mk_bf, SIZE_M * DIM_S);
    transpose_cvt_kernel<<<(K3E * K3E + 255) / 256, 256, 0, stream>>>(W_pt, WptT, K3E, K3E);
    transpose_cvt_kernel<<<(256 * 128 + 255) / 256, 256, 0, stream>>>(W_f, WfT, 256, 128);
    build_wea_kernel<<<(256 * 512 + 255) / 256, 256, 0, stream>>>(W_e, W_a, WeaT);

    big_gemm_cw_kernel<<<ROWS_BIG / 64, 256, 0, stream>>>(
        code, nodes_bf, paths_bf, WptT, b_pt, W_att, b_att, cw);

    softmax_L_kernel<<<BSZ * PLEN, 128, 0, stream>>>(cw);

    codevec_kernel<<<ROWS_SEQ, 384, 0, stream>>>(code, cw, nodes, paths, cvec_bf);

    wsoft_gemm_kernel<<<ROWS_SEQ / 32, 256, 0, stream>>>(q, kemb_bf, Mk_bf, w_ws);

    ea_gemm_kernel<<<ROWS_SEQ / 16, 256, 0, stream>>>(
        r, vemb_bf, cvec_bf, WeaT, b_e, b_a, e_ws, a_ws);

    scan_kernel<<<BSZ, 128, 0, stream>>>(Mv0, w_ws, e_ws, a_ws, read_ws);

    out_gemm_kernel<<<ROWS_SEQ / 16, 256, 0, stream>>>(
        read_ws, q, kemb_bf, WfT, b_f, W_p, b_p, out);
}
